// TemporalGAT_9337258901556
// MI455X (gfx1250) — matrix operand order repaired and checked by reading, not compiled, not run on hardware
//
#include <hip/hip_runtime.h>
#include <hip/hip_bf16.h>

typedef __attribute__((ext_vector_type(16))) _Float16 v16h;
typedef __attribute__((ext_vector_type(8)))  float    v8f;

#define HEADS 4
#define HID 64
#define C1 256            // HEADS*HID
#define NEG_SLOPE 0.2f
#define NGRAPH 64

// ---------- helpers ----------
__device__ __forceinline__ unsigned fenc(float f) {
  unsigned b = __float_as_uint(f);
  return (b & 0x80000000u) ? ~b : (b | 0x80000000u);   // order-preserving float->uint
}
__device__ __forceinline__ float fdec(unsigned u) {
  unsigned b = (u & 0x80000000u) ? (u ^ 0x80000000u) : ~u;
  return __uint_as_float(b);
}
__device__ __forceinline__ float elu1(float v) { return v > 0.f ? v : (expf(v) - 1.f); }

__global__ void k_zero(float* p, int n) {
  int i = blockIdx.x * blockDim.x + threadIdx.x;
  if (i < n) p[i] = 0.f;
}

// fold a_src1/a_dst1 through W1 (layer-1 K=4): av[h*4+k] / av[16+h*4+k]
__global__ void k_fold(const float* W1, const float* a1s, const float* a1d, float* av) {
  int t = threadIdx.x;
  if (t >= 32) return;
  int hk = t & 15;
  int h = hk >> 2, k = hk & 3;
  const float* avec = (t < 16) ? a1s : a1d;
  float s = 0.f;
  for (int c = 0; c < HID; ++c) s += W1[k * C1 + h * HID + c] * avec[h * HID + c];
  av[t] = s;
}

// per-node alpha (layer 1): 4 FMAs per (n,h) via folded vectors
__global__ void k_alpha1(const float* x, const float* av, float* as1, float* ad1, int NT) {
  int idx = blockIdx.x * blockDim.x + threadIdx.x;
  if (idx >= NT * HEADS) return;
  int n = idx >> 2, h = idx & 3;
  float s = 0.f, d = 0.f;
  #pragma unroll
  for (int k = 0; k < 4; ++k) {
    float xv = x[n * 4 + k];
    s += xv * av[h * 4 + k];
    d += xv * av[16 + h * 4 + k];
  }
  as1[idx] = s;
  ad1[idx] = d;
}

// layer-1 edge pass 1: score + segment-max (atomic on encoded uint)
__global__ void k_e1a(const int* ei, int E, int Etot, const float* as1, const float* ad1,
                      float* e1, unsigned* m1) {
  int idx = blockIdx.x * blockDim.x + threadIdx.x;
  if (idx >= Etot * HEADS) return;
  int e = idx >> 2, h = idx & 3;
  int s, d;
  if (e < E) { s = ei[e]; d = ei[E + e]; } else { s = d = e - E; }
  float v = as1[s * 4 + h] + ad1[d * 4 + h];
  v = v > 0.f ? v : NEG_SLOPE * v;
  e1[idx] = v;
  atomicMax(&m1[d * 4 + h], fenc(v));
}

// layer-1 edge pass 2: exp(e - m[dst]), segment-sum
__global__ void k_e1b(const int* ei, int E, int Etot, float* e1, const unsigned* m1, float* dn1) {
  int idx = blockIdx.x * blockDim.x + threadIdx.x;
  if (idx >= Etot * HEADS) return;
  int e = idx >> 2, h = idx & 3;
  int d;
  if (e < E) { d = ei[E + e]; } else { d = e - E; }
  float ex = expf(e1[idx] - fdec(m1[d * 4 + h]));
  e1[idx] = ex;
  atomicAdd(&dn1[d * 4 + h], ex);
}

// layer-1 edge pass 3: recompute h1[src] (K=4) on the fly, scatter msg*alpha
// 256 consecutive threads share one edge -> s,d are wave-uniform, W1/agg accesses coalesced
__global__ void k_e1c(const int* ei, int E, int Etot, const float* x, const float* W1,
                      const float* e1, const float* dn1, float* agg1) {
  int idx = blockIdx.x * blockDim.x + threadIdx.x;
  if (idx >= Etot * C1) return;
  int e = idx >> 8;
  int c256 = idx & 255;
  int h = c256 >> 6;
  int s, d;
  if (e < E) { s = ei[e]; d = ei[E + e]; } else { s = d = e - E; }
  float hv = x[s * 4 + 0] * W1[0 * C1 + c256] + x[s * 4 + 1] * W1[1 * C1 + c256]
           + x[s * 4 + 2] * W1[2 * C1 + c256] + x[s * 4 + 3] * W1[3 * C1 + c256];
  float alpha = e1[e * 4 + h] / dn1[d * 4 + h];
  atomicAdd(&agg1[d * C1 + c256], hv * alpha);
}

// bias + ELU -> f16 activations (GEMM input)
__global__ void k_act1(const float* agg1, const float* b1, _Float16* X2h, int n) {
  int i = blockIdx.x * blockDim.x + threadIdx.x;
  if (i >= n) return;
  X2h[i] = (_Float16)elu1(agg1[i] + b1[i & 255]);
}

// Permute W2 [256x64] f32 into WMMA B-fragment order, f16:
//   W2p[((cc*4 + nt)*32 + lane)*16 + i] = W2[(cc*32 + (lane>>4)*8 + (i&7) + (i>>3)*16)*64 + nt*16 + (lane&15)]
// so each lane's 16 B elements for one (k-chunk, n-tile) are 32 contiguous bytes.
__global__ void k_prepB(const float* W2, _Float16* W2p, int n) {
  int idx = blockIdx.x * blockDim.x + threadIdx.x;
  if (idx >= n) return;
  int i    = idx & 15;
  int lane = (idx >> 4) & 31;
  int nt   = (idx >> 9) & 3;
  int cc   = idx >> 11;
  int K = cc * 32 + (lane >> 4) * 8 + (i & 7) + (i >> 3) * 16;
  int N = nt * 16 + (lane & 15);
  W2p[idx] = (_Float16)W2[K * HID + N];
}

// Layer-2 GEMM: [NTx256]f16 * [256x64]f16 -> [NTx64]f32 via v_wmma_f32_16x16x32_f16.
// One 16x16 output tile per wave (4 waves cover N=64). A and B both load as 2x b128
// per K-step straight from global (B pre-permuted, L2-resident).
__global__ void __launch_bounds__(128) k_gemm2(const _Float16* X2h, const _Float16* W2p, float* H2) {
  int wave   = threadIdx.x >> 5;
  int lane   = threadIdx.x & 31;
  int laneLo = lane & 15;
  int laneHi = lane >> 4;
  int rowBase = blockIdx.x * 16;
  int nBase   = wave * 16;
  const uint4* bbase = (const uint4*)W2p;
  v8f acc = {};
  #pragma unroll
  for (int cc = 0; cc < 8; ++cc) {
    // A frag: lane holds row (lane&15), K = cc*32 + laneHi*8 + {0..7, 16..23}
    union { uint4 u4[2]; v16h v; } au;
    const uint4* ap = (const uint4*)(X2h + (size_t)(rowBase + laneLo) * C1 + cc * 32 + laneHi * 8);
    au.u4[0] = ap[0];
    au.u4[1] = ap[2];
    // B frag: contiguous 32B per lane in fragment-order buffer
    union { uint4 u4[2]; v16h v; } bu;
    const uint4* bp = bbase + (size_t)((cc * 4 + wave) * 32 + lane) * 2;
    bu.u4[0] = bp[0];
    bu.u4[1] = bp[1];
    acc = __builtin_amdgcn_wmma_f32_16x16x32_f16(false, au.v, false, bu.v, (short)0, acc, false, false);
  }
  int row = rowBase + laneHi * 8;
  #pragma unroll
  for (int r = 0; r < 8; ++r)
    H2[(size_t)(row + r) * HID + nBase + laneLo] = acc[r];
}

// per-node alpha (layer 2, 1 head)
__global__ void k_alpha2(const float* H2, const float* a2s, const float* a2d,
                         float* as2, float* ad2, int NT) {
  int n = blockIdx.x * blockDim.x + threadIdx.x;
  if (n >= NT) return;
  float s = 0.f, d = 0.f;
  for (int c = 0; c < HID; ++c) {
    float hv = H2[n * HID + c];
    s += hv * a2s[c];
    d += hv * a2d[c];
  }
  as2[n] = s;
  ad2[n] = d;
}

__global__ void k_e2a(const int* ei, int E, int Etot, const float* as2, const float* ad2,
                      float* e2, unsigned* m2) {
  int e = blockIdx.x * blockDim.x + threadIdx.x;
  if (e >= Etot) return;
  int s, d;
  if (e < E) { s = ei[e]; d = ei[E + e]; } else { s = d = e - E; }
  float v = as2[s] + ad2[d];
  v = v > 0.f ? v : NEG_SLOPE * v;
  e2[e] = v;
  atomicMax(&m2[d], fenc(v));
}

__global__ void k_e2b(const int* ei, int E, int Etot, float* e2, const unsigned* m2, float* dn2) {
  int e = blockIdx.x * blockDim.x + threadIdx.x;
  if (e >= Etot) return;
  int d;
  if (e < E) { d = ei[E + e]; } else { d = e - E; }
  float ex = expf(e2[e] - fdec(m2[d]));
  e2[e] = ex;
  atomicAdd(&dn2[d], ex);
}

__global__ void k_e2c(const int* ei, int E, int Etot, const float* H2,
                      const float* e2, const float* dn2, float* agg2) {
  int idx = blockIdx.x * blockDim.x + threadIdx.x;
  if (idx >= Etot * HID) return;
  int e = idx >> 6, c = idx & 63;
  int s, d;
  if (e < E) { s = ei[e]; d = ei[E + e]; } else { s = d = e - E; }
  float alpha = e2[e] / dn2[d];
  atomicAdd(&agg2[d * HID + c], H2[s * HID + c] * alpha);
}

// bias + ELU + per-graph pooling (sums, counts)
__global__ void k_act2pool(const float* agg2, const float* b2, const int* batch,
                           float* sums, float* cnt, int n) {
  int idx = blockIdx.x * blockDim.x + threadIdx.x;
  if (idx >= n) return;
  int node = idx >> 6, c = idx & 63;
  float v = elu1(agg2[idx] + b2[c]);
  int g = batch[node];
  atomicAdd(&sums[g * HID + c], v);
  if (c == 0) atomicAdd(&cnt[g], 1.0f);
}

// sequential GRU over graphs + FC; single block of 192 threads
__global__ void k_gru(const float* sums, const float* cnt,
                      const float* Wih, const float* Whh,
                      const float* bih, const float* bhh,
                      const float* Wfc, const float* bfc, float* out) {
  __shared__ float h[HID], xt[HID], gi[3 * HID], gh[3 * HID], hnew[HID], red[HID];
  int t = threadIdx.x;
  if (t < HID) h[t] = 0.f;
  __syncthreads();
  for (int g = 0; g < NGRAPH; ++g) {
    if (t < HID) {
      float c = cnt[g];
      c = c > 1.f ? c : 1.f;
      xt[t] = sums[g * HID + t] / c;
    }
    __syncthreads();
    {
      float si = bih[t], sh = bhh[t];
      for (int k = 0; k < HID; ++k) {
        si += Wih[t * HID + k] * xt[k];
        sh += Whh[t * HID + k] * h[k];
      }
      gi[t] = si;
      gh[t] = sh;
    }
    __syncthreads();
    if (t < HID) {
      float r  = 1.f / (1.f + expf(-(gi[t] + gh[t])));
      float z  = 1.f / (1.f + expf(-(gi[HID + t] + gh[HID + t])));
      float nn = tanhf(gi[2 * HID + t] + r * gh[2 * HID + t]);
      hnew[t] = (1.f - z) * nn + z * h[t];
    }
    __syncthreads();
    if (t < HID) {
      h[t] = hnew[t];
      red[t] = hnew[t] * Wfc[t];
    }
    __syncthreads();
    if (t == 0) {
      float s = bfc[0];
      for (int k = 0; k < HID; ++k) s += red[k];
      out[g] = s;
    }
    __syncthreads();
  }
}

extern "C" void kernel_launch(void* const* d_in, const int* in_sizes, int n_in,
                              void* d_out, int out_size, void* d_ws, size_t ws_size,
                              hipStream_t stream) {
  const float* x    = (const float*)d_in[0];
  const int*   ei   = (const int*)d_in[1];
  const int*   batch= (const int*)d_in[2];
  const float* W1   = (const float*)d_in[3];
  const float* a1s  = (const float*)d_in[4];
  const float* a1d  = (const float*)d_in[5];
  const float* b1   = (const float*)d_in[6];
  const float* W2   = (const float*)d_in[7];
  const float* a2s  = (const float*)d_in[8];
  const float* a2d  = (const float*)d_in[9];
  const float* b2   = (const float*)d_in[10];
  const float* Wih  = (const float*)d_in[11];
  const float* Whh  = (const float*)d_in[12];
  const float* bih  = (const float*)d_in[13];
  const float* bhh  = (const float*)d_in[14];
  const float* Wfc  = (const float*)d_in[15];
  const float* bfc  = (const float*)d_in[16];
  float* out = (float*)d_out;

  const int NT   = in_sizes[0] / 4;     // 20000 nodes
  const int E    = in_sizes[1] / 2;     // 320000 edges
  const int Etot = E + NT;              // + self-loops

  // ---- workspace layout (all 256B-aligned) ----
  char* ws = (char*)d_ws;
  size_t off = 0;
  auto alloc = [&](size_t bytes) -> char* {
    char* p = ws + off;
    off += (bytes + 255) & ~(size_t)255;
    return p;
  };
  float*    agg1 = (float*)alloc((size_t)NT * C1 * 4);       // reused below for H2/agg2
  _Float16* X2h  = (_Float16*)alloc((size_t)NT * C1 * 2);
  float*    e1   = (float*)alloc((size_t)Etot * HEADS * 4);  // reused as e2
  unsigned* m1   = (unsigned*)alloc((size_t)NT * HEADS * 4); // reused as m2
  float*    dn1  = (float*)alloc((size_t)NT * HEADS * 4);    // reused as dn2
  float*    as1  = (float*)alloc((size_t)NT * HEADS * 4);
  float*    ad1  = (float*)alloc((size_t)NT * HEADS * 4);
  float*    av   = (float*)alloc(32 * 4);
  _Float16* W2p  = (_Float16*)alloc((size_t)C1 * HID * 2);   // fragment-order W2
  float*    as2  = (float*)alloc((size_t)NT * 4);
  float*    ad2  = (float*)alloc((size_t)NT * 4);
  float*    sums = (float*)alloc((size_t)NGRAPH * HID * 4);
  float*    cnt  = (float*)alloc((size_t)NGRAPH * 4);
  // aliases inside agg1 region (agg1 dead after k_act1):
  float* H2   = agg1;                          // NT*64
  float* agg2 = agg1 + (size_t)NT * HID;       // NT*64 (fits in NT*256 region)

  const int BS = 256;
  auto nb = [](long long n, int bs) { return (int)((n + bs - 1) / bs); };

  // ---- Layer 1 ----
  k_fold<<<1, 32, 0, stream>>>(W1, a1s, a1d, av);
  k_alpha1<<<nb((long long)NT * HEADS, BS), BS, 0, stream>>>(x, av, as1, ad1, NT);
  k_zero<<<nb((long long)NT * C1, BS), BS, 0, stream>>>(agg1, NT * C1);
  k_zero<<<nb((long long)NT * HEADS, BS), BS, 0, stream>>>((float*)m1, NT * HEADS);
  k_zero<<<nb((long long)NT * HEADS, BS), BS, 0, stream>>>(dn1, NT * HEADS);
  k_e1a<<<nb((long long)Etot * HEADS, BS), BS, 0, stream>>>(ei, E, Etot, as1, ad1, e1, m1);
  k_e1b<<<nb((long long)Etot * HEADS, BS), BS, 0, stream>>>(ei, E, Etot, e1, m1, dn1);
  k_e1c<<<nb((long long)Etot * C1, BS), BS, 0, stream>>>(ei, E, Etot, x, W1, e1, dn1, agg1);
  k_act1<<<nb((long long)NT * C1, BS), BS, 0, stream>>>(agg1, b1, X2h, NT * C1);

  // ---- Layer 2 GEMM (WMMA) ----
  k_prepB<<<nb(C1 * HID, BS), BS, 0, stream>>>(W2, W2p, C1 * HID);
  k_gemm2<<<NT / 16, 128, 0, stream>>>(X2h, W2p, H2);

  // ---- Layer 2 attention ----
  k_alpha2<<<nb(NT, BS), BS, 0, stream>>>(H2, a2s, a2d, as2, ad2, NT);
  k_zero<<<nb((long long)NT * HID, BS), BS, 0, stream>>>(agg2, NT * HID);
  k_zero<<<nb(NT, BS), BS, 0, stream>>>((float*)m1, NT);
  k_zero<<<nb(NT, BS), BS, 0, stream>>>(dn1, NT);
  k_zero<<<nb(NGRAPH * HID, BS), BS, 0, stream>>>(sums, NGRAPH * HID);
  k_zero<<<1, NGRAPH, 0, stream>>>(cnt, NGRAPH);
  k_e2a<<<nb(Etot, BS), BS, 0, stream>>>(ei, E, Etot, as2, ad2, e1, m1);
  k_e2b<<<nb(Etot, BS), BS, 0, stream>>>(ei, E, Etot, e1, m1, dn1);
  k_e2c<<<nb((long long)Etot * HID, BS), BS, 0, stream>>>(ei, E, Etot, H2, e1, dn1, agg2);

  // ---- pool + GRU + FC ----
  k_act2pool<<<nb((long long)NT * HID, BS), BS, 0, stream>>>(agg2, b2, batch, sums, cnt, NT * HID);
  k_gru<<<1, 3 * HID, 0, stream>>>(sums, cnt, Wih, Whh, bih, bhh, Wfc, bfc, out);
}
